// TransitionDown_32298154066757
// MI455X (gfx1250) — matrix operand order repaired and checked by reading, not compiled, not run on hardware
//
#include <hip/hip_runtime.h>

// ---------------------------------------------------------------------------
// TransitionDown for MI455X (gfx1250, wave32, WMMA)
//   p[32768,3], x[32768,64], W[67,128], gamma/beta[128]
//   out = concat(new_p[8192,3], x_out[8192,128]) flat f32
// ---------------------------------------------------------------------------

typedef __attribute__((ext_vector_type(16))) _Float16 v16h;
typedef __attribute__((ext_vector_type(8)))  float    v8f;

#define N_PTS   32768
#define M_PTS   8192            // N / STRIDE
#define KNN_K   16
#define CIN     64
#define CPAD    96              // (3 + 64) padded to multiple of 32 for WMMA K
#define COUT    128
#define NROWS   (M_PTS * KNN_K) // 131072 rows of the grouped GEMM
#define BN_EPS  1e-5f

// ---------------------------------------------------------------------------
// Kernel 1: furthest point sampling. One 1024-thread workgroup; all points
// live in registers (32 per lane). 8191 dependent steps: distance to last
// picked point, block-wide argmax (first-max tie-break = lowest index).
// ---------------------------------------------------------------------------
__global__ __launch_bounds__(1024)
void fps_kernel(const float* __restrict__ p, float* __restrict__ new_p)
{
    const int tid  = threadIdx.x;
    const int lane = tid & 31;
    const int wave = tid >> 5;      // 0..31

    __shared__ float lp[3];         // last selected point (broadcast)
    __shared__ float wbd[32];
    __shared__ int   wbi[32];

    // Cache the whole cloud in registers: 32 float3 per lane.
    float px[32], py[32], pz[32];
#pragma unroll
    for (int t = 0; t < 32; ++t) {
        const int i = tid + t * 1024;
        px[t] = p[i * 3 + 0];
        py[t] = p[i * 3 + 1];
        pz[t] = p[i * 3 + 2];
    }

    if (tid == 0) {
        new_p[0] = p[0]; new_p[1] = p[1]; new_p[2] = p[2];  // fps_idx[0] = 0
        lp[0] = p[0]; lp[1] = p[1]; lp[2] = p[2];
    }
    __syncthreads();

    for (int j = 1; j < M_PTS; ++j) {
        const float qx = lp[0], qy = lp[1], qz = lp[2];
        float bd = -1.0f;
        int   bi = 0x7fffffff;
#pragma unroll
        for (int t = 0; t < 32; ++t) {
            const float dx = px[t] - qx, dy = py[t] - qy, dz = pz[t] - qz;
            const float d  = dx * dx + dy * dy + dz * dz;
            const int   i  = tid + t * 1024;
            if (d > bd || (d == bd && i < bi)) { bd = d; bi = i; }
        }
        // wave32 argmax reduce
#pragma unroll
        for (int m = 16; m >= 1; m >>= 1) {
            const float od = __shfl_xor(bd, m, 32);
            const int   oi = __shfl_xor(bi, m, 32);
            if (od > bd || (od == bd && oi < bi)) { bd = od; bi = oi; }
        }
        if (lane == 0) { wbd[wave] = bd; wbi[wave] = bi; }
        __syncthreads();
        if (wave == 0) {
            bd = wbd[lane]; bi = wbi[lane];
#pragma unroll
            for (int m = 16; m >= 1; m >>= 1) {
                const float od = __shfl_xor(bd, m, 32);
                const int   oi = __shfl_xor(bi, m, 32);
                if (od > bd || (od == bd && oi < bi)) { bd = od; bi = oi; }
            }
            if (lane == 0) {
                const float nx = p[bi * 3 + 0];
                const float ny = p[bi * 3 + 1];
                const float nz = p[bi * 3 + 2];
                new_p[j * 3 + 0] = nx; new_p[j * 3 + 1] = ny; new_p[j * 3 + 2] = nz;
                lp[0] = nx; lp[1] = ny; lp[2] = nz;
            }
        }
        __syncthreads();
    }
}

// ---------------------------------------------------------------------------
// Kernel 2: brute-force kNN. One query per thread; p tiled through LDS.
// Unsorted top-16 in registers with tracked worst (order is irrelevant:
// downstream BN/ReLU/maxpool are symmetric over the 16 neighbors).
// ---------------------------------------------------------------------------
#define KTS 2048
__global__ __launch_bounds__(256)
void knn_kernel(const float* __restrict__ p, const float* __restrict__ newp,
                int* __restrict__ knn)
{
    __shared__ float sx[KTS], sy[KTS], sz[KTS];
    const int tid = threadIdx.x;
    const int q   = blockIdx.x * 256 + tid;

    const float qx = newp[q * 3 + 0];
    const float qy = newp[q * 3 + 1];
    const float qz = newp[q * 3 + 2];

    float bd[KNN_K];
    int   bidx[KNN_K];
#pragma unroll
    for (int s = 0; s < KNN_K; ++s) { bd[s] = 3.4e38f; bidx[s] = 0; }
    float worst = 3.4e38f;

    for (int base = 0; base < N_PTS; base += KTS) {
        __syncthreads();
        for (int t = tid; t < KTS; t += 256) {
            const int i = base + t;
            sx[t] = p[i * 3 + 0];
            sy[t] = p[i * 3 + 1];
            sz[t] = p[i * 3 + 2];
        }
        __syncthreads();
        for (int t = 0; t < KTS; ++t) {
            const float dx = sx[t] - qx, dy = sy[t] - qy, dz = sz[t] - qz;
            const float d  = dx * dx + dy * dy + dz * dz;
            if (d < worst) {
                int   ws = 0;
                float wv = bd[0];
#pragma unroll
                for (int s = 1; s < KNN_K; ++s)
                    if (bd[s] > wv) { wv = bd[s]; ws = s; }
#pragma unroll
                for (int s = 0; s < KNN_K; ++s)
                    if (s == ws) { bd[s] = d; bidx[s] = base + t; }
                wv = bd[0];
#pragma unroll
                for (int s = 1; s < KNN_K; ++s) wv = fmaxf(wv, bd[s]);
                worst = wv;
            }
        }
    }
#pragma unroll
    for (int s = 0; s < KNN_K; ++s) knn[q * KNN_K + s] = bidx[s];
}

// ---------------------------------------------------------------------------
// Kernels 3/5: grouped-gather GEMM via v_wmma_f32_16x16x32_f16.
//   256 threads = 8 waves; each wave owns 16 rows (= the 16 neighbors of one
//   sample m) and loops over 8 column tiles, 3 k-blocks of 32 each.
//
//   A and B tiles are staged in LDS PRE-SWIZZLED into the exact per-lane
//   fragment layout (ISA 7.12.2), so each fragment read is a single aligned
//   32-byte vector load (2x ds_load_b128) instead of 16x ds_load_u16.
//   The three A fragments per wave are ct-invariant and held in registers.
//
//   PASS 1: accumulate per-channel sum / sum-of-squares (LDS then global).
//   PASS 2: recompute tile, apply BN scale/shift, ReLU, max over the 16 rows
//           (M-dimension of the C/D fragment) -> x_out[m, :].
// ---------------------------------------------------------------------------
template <int PASS>
__global__ __launch_bounds__(256)
void gemm_kernel(const float* __restrict__ p, const float* __restrict__ x,
                 const float* __restrict__ W, const int* __restrict__ knn,
                 float* __restrict__ gsum, float* __restrict__ gsq,
                 const float* __restrict__ scale, const float* __restrict__ shift,
                 float* __restrict__ xout)
{
    // Fragment-layout staging buffers (32B-aligned vector elements).
    __shared__ v16h Bfrag[CPAD / 32][COUT / 16][32];  // [kb][ct][lane] : 24 KB
    __shared__ v16h Afrag[8][CPAD / 32][32];          // [wave][kb][lane]: 24 KB
    __shared__ float s_sum[COUT], s_sq[COUT];

    const int tid  = threadIdx.x;
    const int wave = tid >> 5;
    const int lane = tid & 31;
    const int half = lane >> 4;   // lane group (0: lanes 0-15, 1: lanes 16-31)
    const int hl   = lane & 15;

    if (PASS == 1 && tid < COUT) { s_sum[tid] = 0.0f; s_sq[tid] = 0.0f; }

    // ---- Stage W -> B fragments, f16, zero-padded K rows 67..95.
    // B 32x16 layout: lane -> N = lane%16; element e -> K = (e/8)*16 + (lane/16)*8 + e%8.
    for (int t = tid; t < (CPAD / 32) * (COUT / 16) * 32 * 16; t += 256) {
        const int e  = t & 15;
        const int l  = (t >> 4) & 31;
        const int ct = (t >> 9) & 7;
        const int kb = t >> 12;
        const int K  = kb * 32 + (e >> 3) * 16 + (l >> 4) * 8 + (e & 7);
        const int c  = ct * 16 + (l & 15);
        Bfrag[kb][ct][l][e] = (K < 67) ? (_Float16)W[K * COUT + c] : (_Float16)0.0f;
    }

    // ---- Stage grouped rows -> A fragments for this wave's 16 rows.
    // A 16x32 layout: lane -> M = lane%16, half = lane/16;
    //   element e (vr = e/2): K = (vr/4)*16 + half*8 + (vr%4)*2 + (e%2).
    // Inverse: given kk in [0,32): grp=kk/16, rem=kk%16, hlf=rem/8, r2=rem%8,
    //   vr = grp*4 + r2/2, e = 2*vr + r2%2, dest lane = hlf*16 + row.
    const int wgBase = blockIdx.x * 128;
    for (int t = lane; t < 16 * CPAD; t += 32) {
        const int row = t / CPAD, c = t % CPAD;
        const int r  = wgBase + wave * 16 + row;
        const int jn = knn[r];
        float v;
        if (c < 3)       v = p[jn * 3 + c];
        else if (c < 67) v = x[jn * CIN + (c - 3)];
        else             v = 0.0f;
        const int kb  = c / 32;
        const int kk  = c % 32;
        const int grp = kk >> 4;
        const int rem = kk & 15;
        const int hlf = rem >> 3;
        const int r2  = rem & 7;
        const int vr  = grp * 4 + (r2 >> 1);
        const int e   = vr * 2 + (r2 & 1);
        Afrag[wave][kb][hlf * 16 + row][e] = (_Float16)v;
    }
    __syncthreads();

    // ct-invariant A fragments -> registers (3 x 8 VGPRs).
    const v16h a0 = Afrag[wave][0][lane];
    const v16h a1 = Afrag[wave][1][lane];
    const v16h a2 = Afrag[wave][2][lane];

    for (int ct = 0; ct < COUT / 16; ++ct) {
        v8f acc = {};
        acc = __builtin_amdgcn_wmma_f32_16x16x32_f16(
            false, a0, false, Bfrag[0][ct][lane], (short)0, acc, false, false);
        acc = __builtin_amdgcn_wmma_f32_16x16x32_f16(
            false, a1, false, Bfrag[1][ct][lane], (short)0, acc, false, false);
        acc = __builtin_amdgcn_wmma_f32_16x16x32_f16(
            false, a2, false, Bfrag[2][ct][lane], (short)0, acc, false, false);

        if (PASS == 1) {
            // C/D layout: lane -> N=lane%16, VGPR v -> M = v + 8*(lane/16).
            float s = 0.0f, ss = 0.0f;
#pragma unroll
            for (int vr = 0; vr < 8; ++vr) { const float h = acc[vr]; s += h; ss += h * h; }
            s  += __shfl_xor(s, 16, 32);   // fold the two M-halves
            ss += __shfl_xor(ss, 16, 32);
            if (half == 0) {
                atomicAdd(&s_sum[ct * 16 + hl], s);
                atomicAdd(&s_sq [ct * 16 + hl], ss);
            }
        } else {
            const int   d  = ct * 16 + hl;
            const float sc = scale[d];
            const float sh = shift[d];
            float mx = -3.4e38f;
#pragma unroll
            for (int vr = 0; vr < 8; ++vr) {
                const float h = fmaxf(acc[vr] * sc + sh, 0.0f);  // BN + ReLU
                mx = fmaxf(mx, h);
            }
            mx = fmaxf(mx, __shfl_xor(mx, 16, 32));              // max over 16 rows
            if (half == 0) {
                const int m = blockIdx.x * 8 + wave;
                xout[m * COUT + d] = mx;
            }
        }
    }

    if (PASS == 1) {
        __syncthreads();
        if (tid < COUT) {
            atomicAdd(&gsum[tid], s_sum[tid]);
            atomicAdd(&gsq [tid], s_sq[tid]);
        }
    }
}

// ---------------------------------------------------------------------------
// Kernel 4: fold sums into BN scale/shift per channel.
// ---------------------------------------------------------------------------
__global__ void stats_kernel(const float* __restrict__ gsum, const float* __restrict__ gsq,
                             const float* __restrict__ gamma, const float* __restrict__ beta,
                             float* __restrict__ scale, float* __restrict__ shift)
{
    const int d = threadIdx.x;
    if (d < COUT) {
        const float inv_n = 1.0f / (float)NROWS;
        const float mean  = gsum[d] * inv_n;
        const float var   = gsq[d] * inv_n - mean * mean;
        const float sc    = gamma[d] * rsqrtf(var + BN_EPS);
        scale[d] = sc;
        shift[d] = beta[d] - mean * sc;
    }
}

__global__ void zero_kernel(float* __restrict__ buf, int n)
{
    const int i = blockIdx.x * blockDim.x + threadIdx.x;
    if (i < n) buf[i] = 0.0f;
}

// ---------------------------------------------------------------------------
// Host launcher
// ---------------------------------------------------------------------------
extern "C" void kernel_launch(void* const* d_in, const int* in_sizes, int n_in,
                              void* d_out, int out_size, void* d_ws, size_t ws_size,
                              hipStream_t stream)
{
    (void)in_sizes; (void)n_in; (void)out_size; (void)ws_size;

    const float* p     = (const float*)d_in[0];
    const float* x     = (const float*)d_in[1];
    //                 d_in[2] = batch offsets (unused)
    const float* W     = (const float*)d_in[3];
    const float* gamma = (const float*)d_in[4];
    const float* beta  = (const float*)d_in[5];
    //                 d_in[6] = stride, d_in[7] = nsample (hardcoded)

    float* out   = (float*)d_out;
    float* new_p = out;                 // [8192,3]
    float* xout  = out + M_PTS * 3;     // [8192,128]

    // Workspace layout: 4*128 floats of stats + 131072 knn indices (~514 KB).
    float* wsf   = (float*)d_ws;
    float* gsum  = wsf;
    float* gsq   = wsf + COUT;
    float* scale = wsf + 2 * COUT;
    float* shift = wsf + 3 * COUT;
    int*   knn   = (int*)(wsf + 4 * COUT);

    fps_kernel<<<1, 1024, 0, stream>>>(p, new_p);
    knn_kernel<<<M_PTS / 256, 256, 0, stream>>>(p, new_p, knn);
    zero_kernel<<<1, 256, 0, stream>>>(gsum, 2 * COUT);
    gemm_kernel<1><<<NROWS / 128, 256, 0, stream>>>(p, x, W, knn, gsum, gsq,
                                                    nullptr, nullptr, nullptr);
    stats_kernel<<<1, 128, 0, stream>>>(gsum, gsq, gamma, beta, scale, shift);
    gemm_kernel<2><<<NROWS / 128, 256, 0, stream>>>(p, x, W, knn, gsum, gsq,
                                                    scale, shift, xout);
}
